// _HFDecoderLayer_5162550689807
// MI455X (gfx1250) — matrix operand order repaired and checked by reading, not compiled, not run on hardware
//
#include <hip/hip_runtime.h>
#include <cstddef>
#include <cstdint>

// ---------------------------------------------------------------------------
// MI455X (gfx1250) decoder layer: RMSNorm -> attention -> residual ->
// RMSNorm -> top-2 MoE -> residual.  All GEMMs via v_wmma_f32_16x16x32_bf16.
// Weights are converted fp32->bf16 once per call (121MB bf16 fits in 192MB L2),
// MoE uses per-expert token gather lists (top-2 of 8 => 4x fewer GEMM FLOPs).
// New this round: A-panel staging uses the Tensor Data Mover
// (tensor_load_to_lds + s_wait_tensorcnt) with descriptor-driven LDS padding,
// and B-panel global reads are prefetched one K-step ahead.
// ---------------------------------------------------------------------------

typedef unsigned short u16;                                   // raw bf16 storage
typedef __bf16  v16bf __attribute__((ext_vector_type(16)));   // WMMA A/B frag
typedef float   v8f   __attribute__((ext_vector_type(8)));    // WMMA C/D frag
typedef unsigned int v4u __attribute__((ext_vector_type(4))); // 16B vector ld/st
typedef unsigned int v4ui __attribute__((ext_vector_type(4)));
typedef int v4i __attribute__((ext_vector_type(4)));
typedef int v8i __attribute__((ext_vector_type(8)));

union FragBF { v4u u[2]; v16bf v; u16 h[16]; };

__device__ __forceinline__ u16 f2bf(float f) {
  unsigned int u = __float_as_uint(f);
  u += 0x7FFFu + ((u >> 16) & 1u);     // round-to-nearest-even
  return (u16)(u >> 16);
}
__device__ __forceinline__ float bf2f(u16 h) {
  return __uint_as_float(((unsigned int)h) << 16);
}
__device__ __forceinline__ v8f wmma_bf16(v16bf a, v16bf b, v8f c) {
  return __builtin_amdgcn_wmma_f32_16x16x32_bf16(false, a, false, b, (short)0, c, false, false);
}

// ---- Tensor Data Mover support (probe-confirmed builtins; guarded) --------
#if defined(__has_builtin)
#if __has_builtin(__builtin_amdgcn_tensor_load_to_lds) && \
    __has_builtin(__builtin_amdgcn_s_wait_tensorcnt)
#define HAVE_TDM 1
#endif
#endif

#ifdef HAVE_TDM
// Issue one 2D bf16 tile load (tileW x tileH elements, row stride rowStride
// elements in global) into LDS at ldsByteOff.  Descriptor layout per CDNA5 ISA
// section 8 (D# group0 128b, group1 256b); groups 2/3 zero (2D tensor).
// padInterval/padAmount pad the LDS destination so rows land with the kernel's
// bank-conflict-free stride.
__device__ __forceinline__ void tdm_load_2d_bf16(
    unsigned ldsByteOff, const void* gsrc,
    unsigned tileW, unsigned tileH,
    unsigned long long rowStride,
    unsigned padInterval, unsigned padAmount, int padEn)
{
  unsigned long long ga = (unsigned long long)(uintptr_t)gsrc;
  v4ui g0;
  g0.x = 1u;                                              // count=1, user D#
  g0.y = ldsByteOff;                                      // lds_addr
  g0.z = (unsigned)(ga & 0xFFFFFFFFu);                    // global_addr[31:0]
  g0.w = (unsigned)((ga >> 32) & 0x01FFFFFFu) | (2u << 30); // addr[56:32]|type=2
  unsigned w0 = (1u << 16)                                // data_size=1 (2B)
              | (padEn ? (1u << 20) : 0u)                 // pad_enable
              | ((padInterval & 7u) << 22)
              | ((padAmount & 127u) << 25);
  v8i g1;
  g1[0] = (int)w0;
  g1[1] = (int)((tileW & 0xFFFFu) << 16);                 // tensor_dim0[15:0]
  g1[2] = (int)((tileW >> 16) | ((tileH & 0xFFFFu) << 16)); // dim0[31:16]|dim1[15:0]
  g1[3] = (int)((tileH >> 16) | ((tileW & 0xFFFFu) << 16)); // dim1[31:16]|tile_dim0
  g1[4] = (int)(tileH & 0xFFFFu);                         // tile_dim1
  g1[5] = (int)(unsigned)(rowStride & 0xFFFFFFFFull);     // dim0_stride[31:0]
  g1[6] = (int)(unsigned)((rowStride >> 32) & 0xFFFFull); // dim0_stride[47:32]
  g1[7] = 0;
  v4i z4 = {0, 0, 0, 0};
#if __clang_major__ >= 23
  v8i z8 = {0, 0, 0, 0, 0, 0, 0, 0};
  __builtin_amdgcn_tensor_load_to_lds(g0, g1, z4, z4, z8, 0);
#else
  __builtin_amdgcn_tensor_load_to_lds(g0, g1, z4, z4, 0);
#endif
}
#endif // HAVE_TDM

// ---------------------- constants ------------------------------------------
#define BATCH 2
#define SEQ   1024
#define TOK   2048        // BATCH*SEQ
#define HID   2048
#define NH    32
#define NKV   8
#define HD    64
#define NE    8
#define FF    1024

// ---------------------- generic WMMA GEMM ----------------------------------
// C[M,N](f32) = gather(A)[M,K](bf16) @ B[K,N](bf16)  with optional residual add,
// optional row gather on A, row scatter + scaled accumulate on C.
// Block: 256 threads = 8 waves in a 4(M) x 2(N) grid; wave tile 32x64 =
// 2x4 WMMA 16x16 tiles; BK=64 (two k-steps of 32 per LDS stage).
#define BM 128
#define BN 128
#define BKK 64
#define LDP 72            // padded K-stride (elements) for LDS panels
// TDM LDS padding: rows of 64 bf16 = 32 DWORDs (pad_interval enc 4 = 2^(4+1)),
// pad 4 DWORDs = 16B (enc 3 = 4) -> row stride 144B = LDP elements.
#define TDM_PAD_INTERVAL 4
#define TDM_PAD_AMOUNT   3

__global__ __launch_bounds__(256) void gemm_bf16_wmma(
    const u16* __restrict__ A, const u16* __restrict__ Bw,
    float* __restrict__ C, int M, int N, int K,
    const int* __restrict__ rowIdx, const int* __restrict__ cntPtr,
    const float* __restrict__ addsrc, const float* __restrict__ rowScale,
    int gatherA, int scatterC, int accum)
{
  __shared__ __align__(16) u16 sA[BM * LDP];
  __shared__ __align__(16) u16 sB[BN * LDP];   // B stored transposed: [n][k]

  const int Meff = cntPtr ? *cntPtr : M;
  const int m0b = blockIdx.y * BM;
  const int n0b = blockIdx.x * BN;
  if (m0b >= Meff) return;

  const int tid  = threadIdx.x;
  const int w    = tid >> 5;
  const int lane = tid & 31;
  const int wm   = w >> 1;        // 0..3
  const int wn   = w & 1;         // 0..1
  const int lrow = lane & 15;
  const int lhi  = lane >> 4;

  v8f acc[2][4];
  #pragma unroll
  for (int mi = 0; mi < 2; ++mi)
    #pragma unroll
    for (int ni = 0; ni < 4; ++ni)
      acc[mi][ni] = (v8f){0.f,0.f,0.f,0.f,0.f,0.f,0.f,0.f};

  for (int k0 = 0; k0 < K; k0 += BKK) {
    __syncthreads();
    // ---- stage A panel: BM x BK into padded-row LDS ----------------------
    bool stagedByTdm = false;
#ifdef HAVE_TDM
    if (!gatherA) {
      if (w == 0) {
        // one descriptor moves the whole 128x64 bf16 tile; DMA applies the
        // 16B-per-row LDS padding so rows land at stride LDP
        tdm_load_2d_bf16((unsigned)(uintptr_t)&sA[0],
                         A + (size_t)m0b * K + k0,
                         BKK, BM, (unsigned long long)K,
                         TDM_PAD_INTERVAL, TDM_PAD_AMOUNT, 1);
      }
      stagedByTdm = true;
    }
#endif
    if (!stagedByTdm) {
      #pragma unroll
      for (int i = 0; i < 4; ++i) {
        int c   = i * 256 + tid;          // 0..1023
        int row = c >> 3;                 // 0..127
        int kc  = c & 7;                  // 0..7 (x8 bf16)
        int gr  = m0b + row;
        int sr  = (gr < Meff) ? gr : (Meff - 1);
        if (gatherA) sr = rowIdx[sr];
        v4u val = *reinterpret_cast<const v4u*>(A + (size_t)sr * K + k0 + kc * 8);
        *reinterpret_cast<v4u*>(sA + row * LDP + kc * 8) = val;
      }
    }
    // ---- stage B panel transposed: read [k][n] coalesced, write [n][k] ---
    #pragma unroll
    for (int i = 0; i < 4; ++i) {
      int c  = i * 256 + tid;           // 0..1023
      int kk = c >> 4;                  // 0..63
      int nc = c & 15;                  // 0..15 (x8 cols)
      const u16* gp = Bw + (size_t)(k0 + kk) * N + n0b + nc * 8;
      v4u val = *reinterpret_cast<const v4u*>(gp);
      if (k0 + BKK < K)                 // prefetch next K-step panel
        __builtin_prefetch((const void*)(gp + (size_t)BKK * N), 0, 1);
      const u16* hv = reinterpret_cast<const u16*>(&val);
      #pragma unroll
      for (int j = 0; j < 8; ++j) sB[(nc * 8 + j) * LDP + kk] = hv[j];
    }
#ifdef HAVE_TDM
    __builtin_amdgcn_s_wait_tensorcnt(0);
#endif
    __syncthreads();

    #pragma unroll
    for (int s = 0; s < 2; ++s) {       // two WMMA k-steps of 32
      FragBF af[2], bfr[4];
      #pragma unroll
      for (int mi = 0; mi < 2; ++mi) {
        int row  = wm * 32 + mi * 16 + lrow;
        int base = row * 9 + s * 4 + lhi;          // v4u index (LDP*2/16 = 9)
        af[mi].u[0] = reinterpret_cast<const v4u*>(sA)[base];
        af[mi].u[1] = reinterpret_cast<const v4u*>(sA)[base + 2];
      }
      #pragma unroll
      for (int ni = 0; ni < 4; ++ni) {
        int col  = wn * 64 + ni * 16 + lrow;
        int base = col * 9 + s * 4 + lhi;
        bfr[ni].u[0] = reinterpret_cast<const v4u*>(sB)[base];
        bfr[ni].u[1] = reinterpret_cast<const v4u*>(sB)[base + 2];
      }
      #pragma unroll
      for (int mi = 0; mi < 2; ++mi)
        #pragma unroll
        for (int ni = 0; ni < 4; ++ni)
          acc[mi][ni] = wmma_bf16(af[mi].v, bfr[ni].v, acc[mi][ni]);
    }
  }

  // ---- epilogue ----------------------------------------------------------
  #pragma unroll
  for (int mi = 0; mi < 2; ++mi) {
    #pragma unroll
    for (int ni = 0; ni < 4; ++ni) {
      int gc = n0b + wn * 64 + ni * 16 + lrow;
      #pragma unroll
      for (int j = 0; j < 8; ++j) {
        int lr = wm * 32 + mi * 16 + j + lhi * 8;
        int gr = m0b + lr;
        if (gr >= Meff) continue;
        int orow = scatterC ? rowIdx[gr] : gr;
        size_t off = (size_t)orow * N + gc;
        float v = acc[mi][ni][j];
        if (accum) {
          float scl = rowScale ? rowScale[gr] : 1.0f;
          C[off] += scl * v;
        } else {
          if (addsrc) v += addsrc[off];
          C[off] = v;
        }
      }
    }
  }
}

// ---------------------- fp32 -> bf16 convert --------------------------------
__global__ void f32_to_bf16_kernel(const float* __restrict__ src,
                                   u16* __restrict__ dst, int n) {
  int i = blockIdx.x * 256 + threadIdx.x;
  if (i < n) dst[i] = f2bf(src[i]);
}

// ---------------------- RMSNorm (one token per block) -----------------------
__global__ __launch_bounds__(256) void rmsnorm_bf16_kernel(
    const float* __restrict__ x, const float* __restrict__ w,
    u16* __restrict__ out) {
  int t = blockIdx.x;
  const float* xp = x + (size_t)t * HID;
  __shared__ float red[256];
  float ss = 0.f;
  #pragma unroll
  for (int i = 0; i < 8; ++i) {
    float v = xp[threadIdx.x + i * 256];
    ss += v * v;
  }
  red[threadIdx.x] = ss;
  __syncthreads();
  for (int off = 128; off > 0; off >>= 1) {
    if (threadIdx.x < off) red[threadIdx.x] += red[threadIdx.x + off];
    __syncthreads();
  }
  float rs = rsqrtf(red[0] / (float)HID + 1e-6f);
  #pragma unroll
  for (int i = 0; i < 8; ++i) {
    int d = threadIdx.x + i * 256;
    out[(size_t)t * HID + d] = f2bf(w[d] * xp[d] * rs);
  }
}

// ---------------------- RoPE + head-major layout ----------------------------
__global__ void rope_kernel(const float* __restrict__ lin,
                            const float* __restrict__ cosb,
                            const float* __restrict__ sinb,
                            u16* __restrict__ out, int nheads) {
  int i = blockIdx.x * 256 + threadIdx.x;
  int d = i & 63;
  int rem = i >> 6;
  int hh = rem % nheads;
  int t  = rem / nheads;          // 0..TOK-1
  int b  = t >> 10, s = t & 1023;
  float c  = cosb[(size_t)t * 64 + d];
  float sn = sinb[(size_t)t * 64 + d];
  size_t src = (size_t)t * (nheads * 64) + hh * 64;
  float x  = lin[src + d];
  float xr = (d < 32) ? -lin[src + d + 32] : lin[src + d - 32];
  out[(((size_t)b * nheads + hh) * SEQ + s) * 64 + d] = f2bf(x * c + xr * sn);
}

__global__ void v_layout_kernel(const float* __restrict__ lin,
                                u16* __restrict__ out) {
  int i = blockIdx.x * 256 + threadIdx.x;   // TOK*NKV*64 threads
  int d = i & 63;
  int rem = i >> 6;
  int hk = rem % NKV;
  int t  = rem / NKV;
  int b  = t >> 10, s = t & 1023;
  out[(((size_t)b * NKV + hk) * SEQ + s) * 64 + d] =
      f2bf(lin[(size_t)t * (NKV * 64) + hk * 64 + d]);
}

// ---------------------- WMMA causal attention -------------------------------
// One block = (qblock of 32 rows, head, batch). 256 threads = 8 waves.
// Scores (32x1024 f32) + V^T panel (64x1024 bf16) live in the 320KB LDS.
#define SC_LD 1028
#define VT_LD 1032
#define ATTN_SMEM (32*SC_LD*4 + 64*VT_LD*2 + (256+256+32+32)*4)

__global__ __launch_bounds__(256) void attention_kernel(
    const u16* __restrict__ qbf, const u16* __restrict__ kbf,
    const u16* __restrict__ vbf, u16* __restrict__ obf) {
  extern __shared__ char smem[];
  float* sc    = reinterpret_cast<float*>(smem);
  u16*   Vt    = reinterpret_cast<u16*>(smem + 32 * SC_LD * 4);
  float* redm  = reinterpret_cast<float*>(smem + 32 * SC_LD * 4 + 64 * VT_LD * 2);
  float* reds  = redm + 256;
  float* rowmax = reds + 256;
  float* rowsum = rowmax + 32;

  const int qb = blockIdx.x, h = blockIdx.y, b = blockIdx.z;
  const int hk = h >> 2;              // GQA: 32 q heads -> 8 kv heads
  const int qg0 = qb * 32;
  const int qEnd = qg0 + 31;
  const int tid  = threadIdx.x;
  const int w    = tid >> 5;
  const int lane = tid & 31;
  const int lrow = lane & 15;
  const int lhi  = lane >> 4;

  const size_t qbase = ((size_t)b * NH + h) * SEQ * 64;
  const size_t kbase = ((size_t)b * NKV + hk) * SEQ * 64;

  // stage V^T : Vt[d][j]
  #pragma unroll 4
  for (int i = 0; i < 32; ++i) {
    int c = i * 256 + tid;            // chunks of 8 along d
    int j = c >> 3, dc = c & 7;
    v4u val = *reinterpret_cast<const v4u*>(vbf + kbase + (size_t)j * 64 + dc * 8);
    const u16* hv = reinterpret_cast<const u16*>(&val);
    #pragma unroll
    for (int u = 0; u < 8; ++u) Vt[(dc * 8 + u) * VT_LD + j] = hv[u];
  }

  // Q fragments (2 m-subtiles x 2 k-steps) straight from global
  FragBF qf[2][2];
  #pragma unroll
  for (int mi = 0; mi < 2; ++mi)
    #pragma unroll
    for (int s = 0; s < 2; ++s) {
      int qg = qg0 + mi * 16 + lrow;
      const v4u* p = reinterpret_cast<const v4u*>(
          qbf + qbase + (size_t)qg * 64 + s * 32 + lhi * 16);
      qf[mi][s].u[0] = p[0];
      qf[mi][s].u[1] = p[1];
    }

  // scores = Q K^T  (skip fully-masked key tiles)
  for (int tile = w; tile < 64; tile += 8) {
    int n0 = tile * 16;
    if (n0 > qEnd) break;
    #pragma unroll
    for (int mi = 0; mi < 2; ++mi) {
      v8f accv = (v8f){0.f,0.f,0.f,0.f,0.f,0.f,0.f,0.f};
      #pragma unroll
      for (int s = 0; s < 2; ++s) {
        FragBF kf;
        int j = n0 + lrow;
        const v4u* p = reinterpret_cast<const v4u*>(
            kbf + kbase + (size_t)j * 64 + s * 32 + lhi * 16);
        kf.u[0] = p[0];
        kf.u[1] = p[1];
        accv = wmma_bf16(qf[mi][s].v, kf.v, accv);
      }
      #pragma unroll
      for (int jj = 0; jj < 8; ++jj) {
        int r = mi * 16 + jj + lhi * 8;
        sc[r * SC_LD + n0 + lrow] = accv[jj];
      }
    }
  }
  __syncthreads();

  // masked, scaled softmax (probs kept unnormalized; rowsum applied at end)
  {
    int r = tid >> 3, seg = tid & 7;
    int qi = qg0 + r;
    int c0 = seg * 128;
    float lm = -3.0e38f;
    for (int c = c0; c < c0 + 128; ++c) {
      float v = (c <= qi) ? sc[r * SC_LD + c] * 0.125f : -3.0e38f;
      sc[r * SC_LD + c] = v;
      lm = fmaxf(lm, v);
    }
    redm[r * 8 + seg] = lm;
  }
  __syncthreads();
  if (tid < 32) {
    float m = redm[tid * 8];
    #pragma unroll
    for (int g = 1; g < 8; ++g) m = fmaxf(m, redm[tid * 8 + g]);
    rowmax[tid] = m;
  }
  __syncthreads();
  {
    int r = tid >> 3, seg = tid & 7;
    float rm = rowmax[r];
    int c0 = seg * 128;
    float ls = 0.f;
    for (int c = c0; c < c0 + 128; ++c) {
      float v = __expf(sc[r * SC_LD + c] - rm);
      sc[r * SC_LD + c] = v;
      ls += v;
    }
    reds[r * 8 + seg] = ls;
  }
  __syncthreads();
  if (tid < 32) {
    float s = 0.f;
    #pragma unroll
    for (int g = 0; g < 8; ++g) s += reds[tid * 8 + g];
    rowsum[tid] = s;
  }
  __syncthreads();

  // out = P V : one 16x16 tile per wave (2 m x 4 d)
  {
    int mi = w & 1, ni = w >> 1;
    v8f accv = (v8f){0.f,0.f,0.f,0.f,0.f,0.f,0.f,0.f};
    for (int jb = 0; jb <= qEnd; jb += 32) {
      FragBF pf, vf;
      int r  = mi * 16 + lrow;
      int kb = jb + lhi * 16;
      const float* sp = sc + r * SC_LD + kb;
      #pragma unroll
      for (int u = 0; u < 8; ++u) pf.h[u]     = f2bf(sp[u]);
      #pragma unroll
      for (int u = 0; u < 8; ++u) pf.h[8 + u] = f2bf(sp[8 + u]);
      int d = ni * 16 + lrow;
      const v4u* vp = reinterpret_cast<const v4u*>(Vt + d * VT_LD + jb + lhi * 16);
      vf.u[0] = vp[0];
      vf.u[1] = vp[1];
      accv = wmma_bf16(pf.v, vf.v, accv);
    }
    #pragma unroll
    for (int jj = 0; jj < 8; ++jj) {
      int r  = mi * 16 + jj + lhi * 8;
      int sg = qg0 + r;
      float val = accv[jj] / rowsum[r];
      obf[((size_t)(b * SEQ + sg)) * (NH * HD) + h * 64 + ni * 16 + lrow] = f2bf(val);
    }
  }
}

// ---------------------- gating: sigmoid + top-2 of 8 ------------------------
__global__ __launch_bounds__(256) void gate_topk_kernel(
    const u16* __restrict__ xn, const float* __restrict__ gw,
    const float* __restrict__ bias, float* __restrict__ cw) {
  int t = blockIdx.x;
  int w = threadIdx.x >> 5, lane = threadIdx.x & 31;   // wave w -> expert w
  __shared__ float lg[8];
  float acc = 0.f;
  #pragma unroll 4
  for (int i = 0; i < 64; ++i) {
    int d = lane + i * 32;
    acc += bf2f(xn[(size_t)t * HID + d]) * gw[(size_t)d * NE + w];
  }
  #pragma unroll
  for (int off = 16; off > 0; off >>= 1) acc += __shfl_xor(acc, off, 32);
  if (lane == 0) lg[w] = acc;
  __syncthreads();
  if (threadIdx.x == 0) {
    float rw[NE], scv[NE];
    #pragma unroll
    for (int e = 0; e < NE; ++e) {
      rw[e]  = 1.f / (1.f + __expf(-lg[e]));
      scv[e] = rw[e] + bias[e];
    }
    int i1 = 0;
    #pragma unroll
    for (int e = 1; e < NE; ++e) if (scv[e] > scv[i1]) i1 = e;
    int i2 = (i1 == 0) ? 1 : 0;
    #pragma unroll
    for (int e = 0; e < NE; ++e) if (e != i1 && scv[e] > scv[i2]) i2 = e;
    float s = rw[i1] + rw[i2];
    #pragma unroll
    for (int e = 0; e < NE; ++e) cw[(size_t)t * NE + e] = 0.f;
    cw[(size_t)t * NE + i1] = rw[i1] / s;
    cw[(size_t)t * NE + i2] = rw[i2] / s;
  }
}

// ---------------------- per-expert token list build -------------------------
__global__ void build_lists_kernel(const float* __restrict__ cw,
                                   int* __restrict__ idx, float* __restrict__ wgt,
                                   int* __restrict__ cnt) {
  int e = threadIdx.x;
  if (e >= NE) return;
  int c = 0;
  for (int t = 0; t < TOK; ++t) {
    float v = cw[(size_t)t * NE + e];
    if (v > 0.f) { idx[e * TOK + c] = t; wgt[e * TOK + c] = v; ++c; }
  }
  cnt[e] = c;
}

// ---------------------- silu(g1)*g3 -> bf16 ---------------------------------
__global__ void silu_mul_kernel(const float* __restrict__ g1,
                                const float* __restrict__ g3,
                                u16* __restrict__ h, const int* __restrict__ cnt) {
  int i = blockIdx.x * 256 + threadIdx.x;
  int r = i >> 10;                    // FF = 1024
  if (r >= *cnt) return;
  float a = g1[i];
  float s = a / (1.f + __expf(-a));
  h[i] = f2bf(s * g3[i]);
}

// ---------------------------------------------------------------------------
extern "C" void kernel_launch(void* const* d_in, const int* in_sizes, int n_in,
                              void* d_out, int out_size, void* d_ws, size_t ws_size,
                              hipStream_t stream) {
  const float* hidden   = (const float*)d_in[0];
  const float* cosb     = (const float*)d_in[1];
  const float* sinb     = (const float*)d_in[2];
  const float* ln1_w    = (const float*)d_in[3];
  const float* ln2_w    = (const float*)d_in[4];
  const float* wq       = (const float*)d_in[5];
  const float* wk       = (const float*)d_in[6];
  const float* wv       = (const float*)d_in[7];
  const float* wo       = (const float*)d_in[8];
  const float* gate_w   = (const float*)d_in[9];
  const float* corrb    = (const float*)d_in[10];
  const float* w1       = (const float*)d_in[11];
  const float* w2       = (const float*)d_in[12];
  const float* w3       = (const float*)d_in[13];
  float* out = (float*)d_out;

  // ---- workspace carve ----
  char* p = (char*)d_ws;
  auto alloc = [&](size_t bytes) -> char* {
    char* r = p;
    p += (bytes + 255) & ~(size_t)255;
    return r;
  };
  u16* wq_bf   = (u16*)alloc((size_t)HID * NH * HD * 2);
  u16* wk_bf   = (u16*)alloc((size_t)HID * NKV * HD * 2);
  u16* wv_bf   = (u16*)alloc((size_t)HID * NKV * HD * 2);
  u16* wo_bf   = (u16*)alloc((size_t)NH * HD * HID * 2);
  u16* w1_bf   = (u16*)alloc((size_t)NE * HID * FF * 2);
  u16* w3_bf   = (u16*)alloc((size_t)NE * HID * FF * 2);
  u16* w2_bf   = (u16*)alloc((size_t)NE * FF * HID * 2);
  u16* xn1_bf  = (u16*)alloc((size_t)TOK * HID * 2);
  u16* xn2_bf  = (u16*)alloc((size_t)TOK * HID * 2);
  float* qlin  = (float*)alloc((size_t)TOK * NH * HD * 4);
  float* klin  = (float*)alloc((size_t)TOK * NKV * HD * 4);
  float* vlin  = (float*)alloc((size_t)TOK * NKV * HD * 4);
  u16* q_bf    = (u16*)alloc((size_t)TOK * NH * HD * 2);
  u16* k_bf    = (u16*)alloc((size_t)TOK * NKV * HD * 2);
  u16* v_bf    = (u16*)alloc((size_t)TOK * NKV * HD * 2);
  u16* attn_bf = (u16*)alloc((size_t)TOK * NH * HD * 2);
  float* cw    = (float*)alloc((size_t)TOK * NE * 4);
  int*   eidx  = (int*)alloc((size_t)NE * TOK * 4);
  float* ewgt  = (float*)alloc((size_t)NE * TOK * 4);
  int*   ecnt  = (int*)alloc((size_t)NE * 4);
  float* g1    = (float*)alloc((size_t)TOK * FF * 4);
  float* g3    = (float*)alloc((size_t)TOK * FF * 4);
  u16* h_bf    = (u16*)alloc((size_t)TOK * FF * 2);

  auto conv = [&](const float* s, u16* d, size_t n) {
    f32_to_bf16_kernel<<<dim3((unsigned)((n + 255) / 256)), dim3(256), 0, stream>>>(s, d, (int)n);
  };

  // ---- 0) weights -> bf16 (subsequently resident in 192MB L2) ----
  conv(wq, wq_bf, (size_t)HID * NH * HD);
  conv(wk, wk_bf, (size_t)HID * NKV * HD);
  conv(wv, wv_bf, (size_t)HID * NKV * HD);
  conv(wo, wo_bf, (size_t)NH * HD * HID);
  conv(w1, w1_bf, (size_t)NE * HID * FF);
  conv(w3, w3_bf, (size_t)NE * HID * FF);
  conv(w2, w2_bf, (size_t)NE * FF * HID);

  // ---- 1) RMSNorm #1 ----
  rmsnorm_bf16_kernel<<<dim3(TOK), dim3(256), 0, stream>>>(hidden, ln1_w, xn1_bf);

  // ---- 2) QKV projections (WMMA, A panel via TDM) ----
  gemm_bf16_wmma<<<dim3(NH * HD / BN, TOK / BM), dim3(256), 0, stream>>>(
      xn1_bf, wq_bf, qlin, TOK, NH * HD, HID,
      nullptr, nullptr, nullptr, nullptr, 0, 0, 0);
  gemm_bf16_wmma<<<dim3(NKV * HD / BN, TOK / BM), dim3(256), 0, stream>>>(
      xn1_bf, wk_bf, klin, TOK, NKV * HD, HID,
      nullptr, nullptr, nullptr, nullptr, 0, 0, 0);
  gemm_bf16_wmma<<<dim3(NKV * HD / BN, TOK / BM), dim3(256), 0, stream>>>(
      xn1_bf, wv_bf, vlin, TOK, NKV * HD, HID,
      nullptr, nullptr, nullptr, nullptr, 0, 0, 0);

  // ---- 3) RoPE + head-major bf16 layouts ----
  rope_kernel<<<dim3(TOK * NH * HD / 256), dim3(256), 0, stream>>>(
      qlin, cosb, sinb, q_bf, NH);
  rope_kernel<<<dim3(TOK * NKV * HD / 256), dim3(256), 0, stream>>>(
      klin, cosb, sinb, k_bf, NKV);
  v_layout_kernel<<<dim3(TOK * NKV * HD / 256), dim3(256), 0, stream>>>(vlin, v_bf);

  // ---- 4) causal attention (WMMA QK^T and PV; 266KB LDS per block) ----
  attention_kernel<<<dim3(SEQ / 32, NH, BATCH), dim3(256), ATTN_SMEM, stream>>>(
      q_bf, k_bf, v_bf, attn_bf);

  // ---- 5) output projection + residual (d_out = attn @ wo + hidden) ----
  gemm_bf16_wmma<<<dim3(HID / BN, TOK / BM), dim3(256), 0, stream>>>(
      attn_bf, wo_bf, out, TOK, HID, NH * HD,
      nullptr, nullptr, hidden, nullptr, 0, 0, 0);

  // ---- 6) RMSNorm #2 ----
  rmsnorm_bf16_kernel<<<dim3(TOK), dim3(256), 0, stream>>>(out, ln2_w, xn2_bf);

  // ---- 7) gating (sigmoid + corr-bias top-2) + per-expert token lists ----
  gate_topk_kernel<<<dim3(TOK), dim3(256), 0, stream>>>(xn2_bf, gate_w, corrb, cw);
  build_lists_kernel<<<dim3(1), dim3(32), 0, stream>>>(cw, eidx, ewgt, ecnt);

  // ---- 8) MoE: gathered GEMMs per expert; scatter-accumulate into d_out ----
  for (int e = 0; e < NE; ++e) {
    const u16* w1e = w1_bf + (size_t)e * HID * FF;
    const u16* w3e = w3_bf + (size_t)e * HID * FF;
    const u16* w2e = w2_bf + (size_t)e * FF * HID;
    const int* idx = eidx + (size_t)e * TOK;
    const float* wgt = ewgt + (size_t)e * TOK;
    const int* cnt = ecnt + e;

    gemm_bf16_wmma<<<dim3(FF / BN, TOK / BM), dim3(256), 0, stream>>>(
        xn2_bf, w1e, g1, TOK, FF, HID,
        idx, cnt, nullptr, nullptr, 1, 0, 0);
    gemm_bf16_wmma<<<dim3(FF / BN, TOK / BM), dim3(256), 0, stream>>>(
        xn2_bf, w3e, g3, TOK, FF, HID,
        idx, cnt, nullptr, nullptr, 1, 0, 0);
    silu_mul_kernel<<<dim3(TOK * FF / 256), dim3(256), 0, stream>>>(g1, g3, h_bf, cnt);
    gemm_bf16_wmma<<<dim3(HID / BN, TOK / BM), dim3(256), 0, stream>>>(
        h_bf, w2e, out, TOK, HID, FF,
        idx, cnt, nullptr, wgt, 0, 1, 1);
  }
  (void)n_in; (void)in_sizes; (void)out_size; (void)ws_size;
}
